// DiM_LLM_62861141344775
// MI455X (gfx1250) — matrix operand order repaired and checked by reading, not compiled, not run on hardware
//
#include <hip/hip_runtime.h>
#include <hip/hip_bf16.h>
#include <math.h>

// Problem constants (from reference)
#define NB_L 8
#define C_B  2
#define C_L  512
#define C_D  768
#define C_DI 1536
#define C_DS 16
#define C_DC 4
#define C_DR 48
#define C_V  32000
#define C_XW (C_DR + 2*C_DS)   // 80

typedef __attribute__((ext_vector_type(16))) __bf16 v16bf;
typedef __attribute__((ext_vector_type(8)))  __bf16 v8bf;
typedef __attribute__((ext_vector_type(2)))  __bf16 v2bf;
typedef __attribute__((ext_vector_type(2)))  float  v2f;
typedef __attribute__((ext_vector_type(8)))  float  v8f;

// Packed f32x2 -> bf16x2 (hardware v_cvt_pk_bf16_f32 on gfx1250)
__device__ __forceinline__ unsigned int pack_bf16(float x, float y) {
  union { v2bf v; unsigned int u; } cvt;
  v2f t; t[0] = x; t[1] = y;
  cvt.v = __builtin_convertvector(t, v2bf);
  return cvt.u;
}
__device__ __forceinline__ float silu_f(float x) { return x / (1.f + __expf(-x)); }

// ---------------------------------------------------------------------------
// Generic bf16-WMMA GEMM: C[M,N] = A[M,K](lda) * B[K,N](ldb=N) + bias + add
// Block tile 128x128, KT=32, 256 threads = 8 waves, wave tile 32x64.
// Software-pipelined: next K-tile global loads overlap current tile's WMMAs.
// ---------------------------------------------------------------------------
__global__ __launch_bounds__(256)
void gemm_bf16_wmma(const float* __restrict__ A, const float* __restrict__ Bm,
                    const float* __restrict__ bias, const float* __restrict__ add,
                    float* __restrict__ C, int M, int N, int K, int lda) {
  __shared__ alignas(16) unsigned short lA[128][32];  // [m][k] bf16
  __shared__ alignas(16) unsigned short lB[128][32];  // [n][k] bf16 (transposed)

  const int tid  = threadIdx.x;
  const int lane = tid & 31;
  const int wid  = tid >> 5;
  const int l16  = lane & 15;
  const int hi16 = lane >> 4;          // 0: lanes 0-15, 1: lanes 16-31
  const int wm   = wid >> 1;           // 0..3 -> 32-row strip
  const int wn   = wid & 1;            // 0..1 -> 64-col strip
  const int m0   = blockIdx.y * 128;
  const int n0   = blockIdx.x * 128;

  const bool aMfull = (m0 + 128 <= M);
  const bool bNfull = (n0 + 128 <= N);

  v8f acc[2][4];
#pragma unroll
  for (int i = 0; i < 2; ++i)
#pragma unroll
    for (int j = 0; j < 4; ++j) acc[i][j] = (v8f){0,0,0,0,0,0,0,0};

  float2 ra[8];          // A tile staged in regs (f32 pairs along k)
  float  rb0[8], rb1[8]; // B tile staged in regs (k, k+1 for one n)

  // ---- register-stage one K tile (k0) ----
  auto load_tiles = [&](int k0) {
    const bool kfull = (k0 + 32 <= K);
    if (aMfull && kfull) {
#pragma unroll
      for (int i = 0; i < 8; ++i) {
        int e = tid + i * 256;
        int m = e >> 4, kp = (e & 15) << 1;
        ra[i] = *reinterpret_cast<const float2*>(A + (size_t)(m0 + m) * lda + k0 + kp);
      }
    } else {
#pragma unroll
      for (int i = 0; i < 8; ++i) {
        int e = tid + i * 256;
        int m = e >> 4, kp = (e & 15) << 1;
        int gm = m0 + m, gk = k0 + kp;
        float f0 = 0.f, f1 = 0.f;
        if (gm < M && gk < K) {
          const float* p = A + (size_t)gm * lda + gk;
          f0 = p[0];
          if (gk + 1 < K) f1 = p[1];
        }
        ra[i] = make_float2(f0, f1);
      }
    }
    if (bNfull && kfull) {
#pragma unroll
      for (int i = 0; i < 8; ++i) {
        int e = tid + i * 256;
        int n = e & 127, kp = (e >> 7) << 1;
        const float* p = Bm + (size_t)(k0 + kp) * N + n0 + n;
        rb0[i] = p[0];
        rb1[i] = p[(size_t)N];
      }
    } else {
#pragma unroll
      for (int i = 0; i < 8; ++i) {
        int e = tid + i * 256;
        int n = e & 127, kp = (e >> 7) << 1;
        int gn = n0 + n, gk = k0 + kp;
        float f0 = 0.f, f1 = 0.f;
        if (gn < N) {
          if (gk     < K) f0 = Bm[(size_t)gk * N + gn];
          if (gk + 1 < K) f1 = Bm[(size_t)(gk + 1) * N + gn];
        }
        rb0[i] = f0; rb1[i] = f1;
      }
    }
    // prefetch one more K step ahead (global_prefetch_b8; speculative is safe)
    if (aMfull && k0 + 32 < K)
      __builtin_prefetch(A + (size_t)(m0 + (tid >> 1)) * lda + k0 + 32, 0, 1);
  };

  // ---- convert staged regs to bf16 (v_cvt_pk_bf16_f32) and store to LDS ----
  auto store_lds = [&]() {
#pragma unroll
    for (int i = 0; i < 8; ++i) {
      int e = tid + i * 256;
      int m = e >> 4, kp = (e & 15) << 1;
      *reinterpret_cast<unsigned int*>(&lA[m][kp]) = pack_bf16(ra[i].x, ra[i].y);
    }
#pragma unroll
    for (int i = 0; i < 8; ++i) {
      int e = tid + i * 256;
      int n = e & 127, kp = (e >> 7) << 1;
      *reinterpret_cast<unsigned int*>(&lB[n][kp]) = pack_bf16(rb0[i], rb1[i]);
    }
  };

  load_tiles(0);

  for (int k0 = 0; k0 < K; k0 += 32) {
    __syncthreads();                  // previous tile's frag loads complete
    store_lds();
    __syncthreads();                  // tile visible to all waves

    if (k0 + 32 < K) load_tiles(k0 + 32);   // overlap global latency with WMMAs

    // ---- load fragments per ISA 16-bit A/B layouts ----
    v16bf af[2], bfm[4];
#pragma unroll
    for (int mi = 0; mi < 2; ++mi) {
      const unsigned short* ar = &lA[wm * 32 + mi * 16 + l16][0];
      const int kb = hi16 ? 8 : 0;    // lanes0-15: k 0..7 & 16..23; lanes16-31: 8..15 & 24..31
      v8bf lo = *reinterpret_cast<const v8bf*>(ar + kb);
      v8bf hi = *reinterpret_cast<const v8bf*>(ar + kb + 16);
#pragma unroll
      for (int q = 0; q < 8; ++q) { af[mi][q] = lo[q]; af[mi][q + 8] = hi[q]; }
    }
#pragma unroll
    for (int ni = 0; ni < 4; ++ni) {
      const unsigned short* br = &lB[wn * 64 + ni * 16 + l16][0];
      const int kb = hi16 ? 8 : 0;    // lanes0-15: k 0..7 & 16..23; lanes16-31: 8..15 & 24..31
      v8bf lo = *reinterpret_cast<const v8bf*>(br + kb);
      v8bf hi = *reinterpret_cast<const v8bf*>(br + kb + 16);
#pragma unroll
      for (int q = 0; q < 8; ++q) { bfm[ni][q] = lo[q]; bfm[ni][q + 8] = hi[q]; }
    }
    // ---- 8 WMMAs per K step ----
#pragma unroll
    for (int mi = 0; mi < 2; ++mi)
#pragma unroll
      for (int ni = 0; ni < 4; ++ni)
        acc[mi][ni] = __builtin_amdgcn_wmma_f32_16x16x32_bf16(
            false, af[mi], false, bfm[ni], (short)0, acc[mi][ni], false, false);
  }

  // ---- epilogue: C layout (VGPR r: lanes0-15 M=r, lanes16-31 M=8+r) ----
#pragma unroll
  for (int mi = 0; mi < 2; ++mi) {
#pragma unroll
    for (int ni = 0; ni < 4; ++ni) {
      int gn = n0 + wn * 64 + ni * 16 + l16;
      if (gn >= N) continue;
      float bv = bias ? bias[gn] : 0.f;
      int gmb = m0 + wm * 32 + mi * 16 + (hi16 ? 8 : 0);
#pragma unroll
      for (int r = 0; r < 8; ++r) {
        int gm = gmb + r;
        if (gm < M) {
          float v = acc[mi][ni][r] + bv;
          if (add) v += add[(size_t)gm * N + gn];
          C[(size_t)gm * N + gn] = v;
        }
      }
    }
  }
}

// ---------------------------------------------------------------------------
// t-embedding MLP: temb[b,:] = silu(t*tw1+tb1) @ tw2 + tb2
// ---------------------------------------------------------------------------
__global__ __launch_bounds__(256)
void temb_kernel(const float* __restrict__ t, const float* __restrict__ tw1,
                 const float* __restrict__ tb1, const float* __restrict__ tw2,
                 const float* __restrict__ tb2, float* __restrict__ temb) {
  __shared__ float u[C_D];
  const int b = blockIdx.y;
  const float tv = t[b];
  for (int i = threadIdx.x; i < C_D; i += 256) u[i] = silu_f(tv * tw1[i] + tb1[i]);
  __syncthreads();
  const int d = blockIdx.x * 256 + threadIdx.x;
  float s = tb2[d];
  for (int i = 0; i < C_D; ++i) s += u[i] * tw2[(size_t)i * C_D + d];
  temb[(size_t)b * C_D + d] = s;
}

// Token gather
__global__ void embed_kernel(const int* __restrict__ ids, const float* __restrict__ emb,
                             float* __restrict__ h) {
  int idx = blockIdx.x * 256 + threadIdx.x;       // B*L*D
  int d = idx % C_D;
  int tk = idx / C_D;
  h[idx] = emb[(size_t)ids[tk] * C_D + d];
}

// Fused LayerNorm with optional per-batch residual (t_emb)
__global__ __launch_bounds__(256)
void ln_kernel(const float* __restrict__ x, const float* __restrict__ temb,
               const float* __restrict__ g, const float* __restrict__ bb,
               float* __restrict__ out) {
  __shared__ float red[256];
  const int tok = blockIdx.x;          // 0..B*L-1
  const int b   = tok / C_L;
  const int tid = threadIdx.x;
  float v[3];
#pragma unroll
  for (int i = 0; i < 3; ++i) {
    int d = tid + i * 256;
    float val = x[(size_t)tok * C_D + d];
    if (temb) val += temb[(size_t)b * C_D + d];
    v[i] = val;
  }
  red[tid] = v[0] + v[1] + v[2];
  __syncthreads();
  for (int st = 128; st > 0; st >>= 1) { if (tid < st) red[tid] += red[tid + st]; __syncthreads(); }
  const float mean = red[0] / C_D;
  __syncthreads();
  float sq = 0.f;
#pragma unroll
  for (int i = 0; i < 3; ++i) { float d0 = v[i] - mean; sq += d0 * d0; }
  red[tid] = sq;
  __syncthreads();
  for (int st = 128; st > 0; st >>= 1) { if (tid < st) red[tid] += red[tid + st]; __syncthreads(); }
  const float rstd = rsqrtf(red[0] / C_D + 1e-5f);
#pragma unroll
  for (int i = 0; i < 3; ++i) {
    int d = tid + i * 256;
    out[(size_t)tok * C_D + d] = (v[i] - mean) * rstd * g[d] + bb[d];
  }
}

// Causal depthwise conv (DC=4) over x = xz[..., :DI], + bias, then SiLU
__global__ __launch_bounds__(256)
void conv_silu_kernel(const float* __restrict__ xz, const float* __restrict__ w,
                      const float* __restrict__ cb, float* __restrict__ xc) {
  int idx = blockIdx.x * 256 + threadIdx.x;       // B*L*DI
  int c = idx % C_DI;
  int l = (idx / C_DI) % C_L;
  int b = idx / (C_DI * C_L);
  float s = cb[c];
#pragma unroll
  for (int k = 0; k < C_DC; ++k) {
    int ll = l - (C_DC - 1) + k;
    if (ll >= 0) s += w[c * C_DC + k] * xz[((size_t)(b * C_L + ll)) * (2 * C_DI) + c];
  }
  xc[idx] = silu_f(s);
}

// dt = softplus(dt_proj + b_dt)
__global__ void dt_softplus_kernel(float* __restrict__ dt, const float* __restrict__ bdt) {
  int idx = blockIdx.x * 256 + threadIdx.x;
  int c = idx % C_DI;
  float p = dt[idx] + bdt[c];
  dt[idx] = (p > 20.f) ? p : log1pf(__expf(p));
}

// Sequential SSM scan, parallel over (b, channel). B_t/C_t broadcast via LDS.
__global__ __launch_bounds__(256)
void ssm_scan_kernel(const float* __restrict__ xdbl, const float* __restrict__ dt,
                     const float* __restrict__ xc, const float* __restrict__ Alog,
                     float* __restrict__ y) {
  __shared__ float sB[C_DS], sC[C_DS];
  const int c = blockIdx.x * 256 + threadIdx.x;
  const int b = blockIdx.y;
  float Ac[C_DS], st[C_DS];
#pragma unroll
  for (int j = 0; j < C_DS; ++j) { Ac[j] = -__expf(Alog[(size_t)c * C_DS + j]); st[j] = 0.f; }
  for (int t = 0; t < C_L; ++t) {
    __syncthreads();
    if (threadIdx.x < C_DS)
      sB[threadIdx.x] = xdbl[((size_t)(b * C_L + t)) * C_XW + C_DR + threadIdx.x];
    else if (threadIdx.x < 2 * C_DS)
      sC[threadIdx.x - C_DS] = xdbl[((size_t)(b * C_L + t)) * C_XW + C_DR + C_DS + (threadIdx.x - C_DS)];
    __syncthreads();
    const size_t base = ((size_t)(b * C_L + t)) * C_DI + c;
    const float dtv = dt[base];
    const float dx  = dtv * xc[base];
    float yv = 0.f;
#pragma unroll
    for (int j = 0; j < C_DS; ++j) {
      st[j] = st[j] * __expf(dtv * Ac[j]) + dx * sB[j];
      yv += st[j] * sC[j];
    }
    y[base] = yv;
  }
}

// y = (y + Dp*x) * silu(z),  z = xz[..., DI:]
__global__ void gate_kernel(float* __restrict__ y, const float* __restrict__ xc,
                            const float* __restrict__ xz, const float* __restrict__ Dp) {
  int idx = blockIdx.x * 256 + threadIdx.x;
  int c = idx % C_DI;
  int tk = idx / C_DI;
  float z = xz[(size_t)tk * (2 * C_DI) + C_DI + c];
  y[idx] = (y[idx] + Dp[c] * xc[idx]) * silu_f(z);
}

// ---------------------------------------------------------------------------
extern "C" void kernel_launch(void* const* d_in, const int* in_sizes, int n_in,
                              void* d_out, int out_size, void* d_ws, size_t ws_size,
                              hipStream_t stream) {
  (void)in_sizes; (void)n_in; (void)out_size; (void)ws_size;
  const int*   ids    = (const int*)  d_in[0];
  const float* tnorm  = (const float*)d_in[1];
  const float* tokemb = (const float*)d_in[2];
  const float* tw1    = (const float*)d_in[3];
  const float* tb1    = (const float*)d_in[4];
  const float* tw2    = (const float*)d_in[5];
  const float* tb2    = (const float*)d_in[6];
  const float* ln_g   = (const float*)d_in[7];
  const float* ln_b   = (const float*)d_in[8];
  const float* W_in   = (const float*)d_in[9];
  const float* b_in   = (const float*)d_in[10];
  const float* conv_w = (const float*)d_in[11];
  const float* conv_b = (const float*)d_in[12];
  const float* W_x    = (const float*)d_in[13];
  const float* W_dt   = (const float*)d_in[14];
  const float* b_dt   = (const float*)d_in[15];
  const float* A_log  = (const float*)d_in[16];
  const float* D_p    = (const float*)d_in[17];
  const float* W_out  = (const float*)d_in[18];
  const float* b_out  = (const float*)d_in[19];
  const float* fn_g   = (const float*)d_in[20];
  const float* fn_b   = (const float*)d_in[21];
  const float* W_head = (const float*)d_in[22];
  const float* b_head = (const float*)d_in[23];
  float* out = (float*)d_out;

  float* ws   = (float*)d_ws;
  float* temb = ws;  ws += (size_t)C_B * C_D;
  float* h    = ws;  ws += (size_t)C_B * C_L * C_D;
  float* xz   = ws;  ws += (size_t)C_B * C_L * 2 * C_DI;
  float* xc   = ws;  ws += (size_t)C_B * C_L * C_DI;
  float* xdbl = ws;  ws += (size_t)C_B * C_L * C_XW;
  float* dtb  = ws;  ws += (size_t)C_B * C_L * C_DI;
  float* yb   = ws;  ws += (size_t)C_B * C_L * C_DI;

  const int M = C_B * C_L;  // 1024 tokens

  auto gemm = [&](const float* Ap, const float* Bp, const float* biasp,
                  const float* addp, float* Cp, int Mm, int Nn, int Kk, int lda) {
    dim3 g((unsigned)((Nn + 127) / 128), (unsigned)((Mm + 127) / 128));
    gemm_bf16_wmma<<<g, 256, 0, stream>>>(Ap, Bp, biasp, addp, Cp, Mm, Nn, Kk, lda);
  };

  temb_kernel<<<dim3(C_D / 256, C_B), 256, 0, stream>>>(tnorm, tw1, tb1, tw2, tb2, temb);
  embed_kernel<<<(M * C_D) / 256, 256, 0, stream>>>(ids, tokemb, h);

  for (int l = 0; l < NB_L; ++l) {
    // h = LN(h + t_emb)  (in place)
    ln_kernel<<<M, 256, 0, stream>>>(h, temb, ln_g + (size_t)l * C_D, ln_b + (size_t)l * C_D, h);
    // xz = h @ W_in + b_in     (1024 x 3072, K=768)
    gemm(h, W_in + (size_t)l * C_D * 2 * C_DI, b_in + (size_t)l * 2 * C_DI, nullptr,
         xz, M, 2 * C_DI, C_D, C_D);
    // x = silu(depthwise_conv(x) + cb)
    conv_silu_kernel<<<(M * C_DI) / 256, 256, 0, stream>>>(
        xz, conv_w + (size_t)l * C_DI * C_DC, conv_b + (size_t)l * C_DI, xc);
    // xdbl = x @ W_x           (1024 x 80, K=1536)
    gemm(xc, W_x + (size_t)l * C_DI * C_XW, nullptr, nullptr, xdbl, M, C_XW, C_DI, C_DI);
    // dt_proj = xdbl[:, :48] @ W_dt   (1024 x 1536, K=48, lda=80)
    gemm(xdbl, W_dt + (size_t)l * C_DR * C_DI, nullptr, nullptr, dtb, M, C_DI, C_DR, C_XW);
    dt_softplus_kernel<<<(M * C_DI) / 256, 256, 0, stream>>>(dtb, b_dt + (size_t)l * C_DI);
    // selective-scan recurrence
    ssm_scan_kernel<<<dim3(C_DI / 256, C_B), 256, 0, stream>>>(
        xdbl, dtb, xc, A_log + (size_t)l * C_DI * C_DS, yb);
    // y = (y + Dp*x) * silu(z)
    gate_kernel<<<(M * C_DI) / 256, 256, 0, stream>>>(yb, xc, xz, D_p + (size_t)l * C_DI);
    // h = h + y @ W_out + b_out   (residual fused into GEMM epilogue)
    gemm(yb, W_out + (size_t)l * C_DI * C_D, b_out + (size_t)l * C_D, h, h, M, C_D, C_DI, C_DI);
  }

  // final LN + vocab head (1024 x 32000, K=768)
  ln_kernel<<<M, 256, 0, stream>>>(h, nullptr, fn_g, fn_b, h);
  gemm(h, W_head, b_head, nullptr, out, M, C_V, C_D, C_D);
}
